// GLCN_43860206027073
// MI455X (gfx1250) — matrix operand order repaired and checked by reading, not compiled, not run on hardware
//
#include <hip/hip_runtime.h>
#include <hip/hip_bf16.h>

typedef __attribute__((ext_vector_type(16))) _Float16 v16h;
typedef __attribute__((ext_vector_type(8)))  _Float16 v8h;
typedef __attribute__((ext_vector_type(8)))  float    v8f;

#define BB  4
#define NB  256   // N
#define DD  256   // D == OUT
#define LL  128   // L
#define LDH 264   // padded LDS row stride (elements) to avoid bank conflicts

// ---------------------------------------------------------------------------
// Swizzle a KxN fp32 weight into WMMA B-fragment order as f16.
// Flat layout: [ct][ks][lane][h], lane=(hi<<4)|n, K = ks*32 + (h&7) + 8*hi + 16*(h>>3).
// Each wave then fetches its 16-half B fragment with one contiguous 32B load.
// ---------------------------------------------------------------------------
__global__ __launch_bounds__(256) void swizzle_k(const float* __restrict__ W,
                                                 _Float16* __restrict__ Wsw,
                                                 int K, int Nout) {
  int idx = blockIdx.x * 256 + threadIdx.x;
  if (idx >= K * Nout) return;
  int ksteps = K >> 5;
  int h    = idx & 15;
  int lane = (idx >> 4) & 31;
  int rest = idx >> 9;
  int ks = rest % ksteps;
  int ct = rest / ksteps;
  int n = lane & 15, hi = lane >> 4;
  int k = ks * 32 + (h & 7) + hi * 8 + (h >> 3) * 16; // B 32x16 f16 striping
  Wsw[idx] = (_Float16)W[(size_t)k * Nout + ct * 16 + n];
}

// ---------------------------------------------------------------------------
// Generic fp32-in / fp32-out GEMM on v_wmma_f32_16x16x32_f16.
// C[M,Nout] = act(A[M,K] @ W + colBias + rowBias[row%N]).  Wave = 16x16 tile.
// ---------------------------------------------------------------------------
__global__ __launch_bounds__(256) void gemm_f16_wmma(
    const float* __restrict__ A, const _Float16* __restrict__ Wsw,
    float* __restrict__ C, int M, int K, int Nout,
    const float* __restrict__ colBias, const float* __restrict__ rowBias,
    int doRelu) {
  int lane = threadIdx.x & 31, wv = threadIdx.x >> 5;
  int r0 = blockIdx.x * 16;
  int ct = blockIdx.y * 8 + wv;
  int m = lane & 15, hi = lane >> 4;
  int ksteps = K >> 5;
  v8f acc = {};
  for (int ks = 0; ks < ksteps; ++ks) {
    // A fragment: halves 0..7 = K[ks*32+hi*8 ..], halves 8..15 = +16 (contiguous runs)
    const float* ar = A + (size_t)(r0 + m) * K + ks * 32 + hi * 8;
    v16h af;
#pragma unroll
    for (int t = 0; t < 8; ++t) af[t] = (_Float16)ar[t];
#pragma unroll
    for (int t = 0; t < 8; ++t) af[8 + t] = (_Float16)ar[16 + t];
    v16h bf = *(const v16h*)(Wsw + ((size_t)(ct * ksteps + ks) * 32 + lane) * 16);
    acc = __builtin_amdgcn_wmma_f32_16x16x32_f16(false, af, false, bf,
                                                 (short)0, acc, false, false);
  }
  int n = lane & 15;
#pragma unroll
  for (int r = 0; r < 8; ++r) {
    int row = r0 + r + hi * 8;                      // D layout: M = r + 8*hi
    int col = ct * 16 + n;
    float v = acc[r];
    if (colBias) v += colBias[col];
    if (rowBias) v += rowBias[(size_t)(row & (NB - 1)) * Nout + col];
    if (doRelu) v = v > 0.f ? v : 0.f;
    C[(size_t)row * Nout + col] = v;
  }
}

// ---------------------------------------------------------------------------
// x_hat = x @ W_proj + b_proj   (fp32: feeds exp-amplified gl_loss path)
// ---------------------------------------------------------------------------
__global__ __launch_bounds__(128) void xhat_k(const float* __restrict__ x,
                                              const float* __restrict__ Wp,
                                              const float* __restrict__ bp,
                                              float* __restrict__ xhat) {
  int row = blockIdx.x, t = threadIdx.x;
  __shared__ float sx[DD];
  sx[t] = x[(size_t)row * DD + t];
  sx[t + 128] = x[(size_t)row * DD + t + 128];
  __syncthreads();
  float s = bp[t];
  for (int k = 0; k < DD; ++k) s = fmaf(sx[k], Wp[(size_t)k * LL + t], s);
  xhat[(size_t)row * LL + t] = s;
}

// ---------------------------------------------------------------------------
// Graph-learning row kernel: one block per (b,i). Produces soft_adj row,
// adj2 = adj*soft_adj, and per-row loss partials (deterministic reduction).
// ---------------------------------------------------------------------------
__global__ __launch_bounds__(256) void softadj_k(
    const float* __restrict__ xhat, const float* __restrict__ adj,
    const int* __restrict__ box_num, const float* __restrict__ lw,
    float* __restrict__ soft_out, float* __restrict__ adj2,
    float* __restrict__ p1, float* __restrict__ p2) {
  int bi = blockIdx.x; int b = bi >> 8, i = bi & 255;
  int j = threadIdx.x;
  __shared__ float sxi[LL], slw[LL], r1[256], r2[256];
  if (j < LL) { sxi[j] = xhat[(size_t)bi * LL + j]; slw[j] = lw[j]; }
  __syncthreads();
  const float* xj = xhat + (size_t)(b * NB + j) * LL;
  float s = 0.f, q = 0.f, lsum = 0.f;
  for (int d = 0; d < LL; ++d) {
    float diff = sxi[d] - xj[d];
    s = fmaf(fabsf(diff), slw[d], s);
    q = fmaf(diff, diff, q);
    lsum += slw[d];
  }
  int bn = box_num[b];
  if (!(i < bn && j < bn)) s -= lsum;                 // mask*-1 dotted with lw
  float out = s > 0.f ? s : 0.01f * s;                // leaky_relu(0.01)
  r1[j] = out;                                        // row max
  for (int off = 128; off > 0; off >>= 1) { __syncthreads(); if (j < off) r1[j] = fmaxf(r1[j], r1[j + off]); }
  __syncthreads();
  float mx = r1[0];
  __syncthreads();
  float aj = adj[(size_t)bi * NB + j];
  float e = expf(out - mx) * aj;
  r1[j] = e;                                          // row sum
  for (int off = 128; off > 0; off >>= 1) { __syncthreads(); if (j < off) r1[j] += r1[j + off]; }
  __syncthreads();
  float rs = r1[0];
  __syncthreads();
  float sa = e / rs + 1e-10f;
  soft_out[(size_t)bi * NB + j] = sa;
  adj2[(size_t)bi * NB + j] = aj * sa;
  float pn = sqrtf(q + 1e-12f);
  r1[j] = expf(sa + pn);                              // ETA = 1.0
  r2[j] = sa * sa;
  for (int off = 128; off > 0; off >>= 1) { __syncthreads(); if (j < off) { r1[j] += r1[j + off]; r2[j] += r2[j + off]; } }
  __syncthreads();
  if (j == 0) { p1[bi] = r1[0]; p2[bi] = r2[0]; }
}

__global__ __launch_bounds__(256) void gl_final_k(const float* __restrict__ p1,
                                                  const float* __restrict__ p2,
                                                  const int* __restrict__ box_num,
                                                  float* __restrict__ gl) {
  int b = blockIdx.x, t = threadIdx.x;
  __shared__ float r1[256], r2[256];
  r1[t] = p1[b * 256 + t];
  r2[t] = p2[b * 256 + t];
  for (int off = 128; off > 0; off >>= 1) { __syncthreads(); if (t < off) { r1[t] += r1[t + off]; r2[t] += r2[t + off]; } }
  __syncthreads();
  if (t == 0) {
    float bn = (float)box_num[b];
    gl[b] = r1[0] / (bn * bn) + 1e-4f * sqrtf(r2[0]);   // GAMMA = 1e-4
  }
}

// ---------------------------------------------------------------------------
// Layer 0: AH0[b,i,k] = sum_j adj2[b,i,j]*relu(xib0[b,i,k]+xjb0[b,j,k]+rel.Wa)
// H0 never materialized. Thread owns column k; one block per (b,i).
// ---------------------------------------------------------------------------
__global__ __launch_bounds__(256) void gcn0_ah_k(
    const float* __restrict__ xib0, const float* __restrict__ xjb0,
    const float* __restrict__ rel, const float* __restrict__ Wa,
    const float* __restrict__ adj2, float* __restrict__ AH0) {
  int bi = blockIdx.x; int b = bi >> 8;
  int k = threadIdx.x;
  __shared__ float sWa[6 * DD], sXi[DD], sAdj[NB];
  sXi[k] = xib0[(size_t)bi * DD + k];
  sAdj[k] = adj2[(size_t)bi * NB + k];
#pragma unroll
  for (int c = 0; c < 6; ++c) sWa[c * DD + k] = Wa[c * DD + k];
  __syncthreads();
  float acc = 0.f;
  const float* relrow = rel + (size_t)bi * NB * 6;
  for (int j = 0; j < NB; ++j) {
    float a = relrow[j * 6 + 0] * sWa[k]
            + relrow[j * 6 + 1] * sWa[DD + k]
            + relrow[j * 6 + 2] * sWa[2 * DD + k]
            + relrow[j * 6 + 3] * sWa[3 * DD + k]
            + relrow[j * 6 + 4] * sWa[4 * DD + k]
            + relrow[j * 6 + 5] * sWa[5 * DD + k];
    float h = sXi[k] + xjb0[(size_t)(b * NB + j) * DD + k] + a;
    h = h > 0.f ? h : 0.f;
    acc = fmaf(sAdj[j], h, acc);
  }
  AH0[(size_t)bi * DD + k] = acc;
}

// ---------------------------------------------------------------------------
// Layer 1 flagship kernel: per (b,i), per 16-row j-tile:
//   p0: async-DMA the xjb1 tile into LDS (overlaps p1/p2, ASYNCcnt-tracked)
//   p1: rebuild H0 tile (f16, LDS)           [recompute beats 270MB spill]
//   p2: alpha1 tile = relu(H0 @ l0_w_alpha)  [v_wmma_f32_16x16x32_f16]
//   p3: H1 = relu(xi1+xjb1+alpha1); AH1 += adj2*H1 (thread owns k)
// ---------------------------------------------------------------------------
__global__ __launch_bounds__(256) void gcn1_fused_k(
    const float* __restrict__ xib0, const float* __restrict__ xjb0,
    const float* __restrict__ rel, const float* __restrict__ Wa,
    const float* __restrict__ xi1, const float* __restrict__ xjb1,
    const _Float16* __restrict__ WswAl, const float* __restrict__ adj2,
    float* __restrict__ AH1) {
  __shared__ _Float16 sAh[16 * LDH];     // H0 tile, f16 (A fragments)
  __shared__ float    sAlp[16 * LDH];    // alpha1 tile, f32
  __shared__ float    sXj1[16 * LDH];    // async-staged xjb1 tile, f32
  __shared__ float    sWa[6 * DD], sXi0[DD], sXi1[DD], sAdj[NB];
  int bi = blockIdx.x; int b = bi >> 8;
  int tid = threadIdx.x;
  int k = tid;
  sXi0[k] = xib0[(size_t)bi * DD + k];
  sXi1[k] = xi1[(size_t)bi * DD + k];
  sAdj[k] = adj2[(size_t)bi * NB + k];
#pragma unroll
  for (int c = 0; c < 6; ++c) sWa[c * DD + k] = Wa[c * DD + k];
  __syncthreads();
  int lane = tid & 31, wv = tid >> 5;
  int m = lane & 15, hi = lane >> 4, n = lane & 15;
  int ty = tid >> 4, tx = tid & 15;
  float acc1 = 0.f;
  for (int jt = 0; jt < 16; ++jt) {
    int j0 = jt * 16;
    {   // phase 0: kick off async global->LDS DMA of this tile's xjb1 rows.
        // 256 threads x 4 x b128 = 16KB; lands before the pre-phase-3 fence.
      unsigned long long gsrc =
          (unsigned long long)(size_t)(xjb1 + (size_t)(b * NB + j0 + ty) * DD + tx * 4);
      unsigned ldsa = (unsigned)(size_t)(sXj1 + ty * LDH + tx * 4);
#pragma unroll
      for (int q = 0; q < 4; ++q) {
        asm volatile("global_load_async_to_lds_b128 %0, %1, off"
                     :: "v"(ldsa + (unsigned)(q * 256)),
                        "v"(gsrc + (unsigned long long)(q * 256))
                     : "memory");
      }
    }
    {   // phase 1: H0 tile rows j0..j0+15 (thread = (row ty, 16 k's at tx*16))
      int jg = j0 + ty;
      const float* relr = rel + ((size_t)bi * NB + jg) * 6;
      float c0 = relr[0], c1 = relr[1], c2 = relr[2];
      float c3 = relr[3], c4 = relr[4], c5 = relr[5];
      const float* xjr = xjb0 + (size_t)(b * NB + jg) * DD;
      if (jt + 1 < 16)
        __builtin_prefetch(xjb0 + (size_t)(b * NB + jg + 16) * DD + tx * 16, 0, 0);
#pragma unroll
      for (int t = 0; t < 16; ++t) {
        int kk = tx * 16 + t;
        float a = c0 * sWa[kk] + c1 * sWa[DD + kk] + c2 * sWa[2 * DD + kk]
                + c3 * sWa[3 * DD + kk] + c4 * sWa[4 * DD + kk] + c5 * sWa[5 * DD + kk];
        float h = sXi0[kk] + xjr[kk] + a;
        sAh[ty * LDH + kk] = (_Float16)(h > 0.f ? h : 0.f);
      }
    }
    __syncthreads();
    // phase 2: alpha1 tile via WMMA; 8 waves x 2 col-tiles cover 256 cols
#pragma unroll
    for (int cti = 0; cti < 2; ++cti) {
      int ct = wv + cti * 8;
      v8f acc = {};
#pragma unroll
      for (int ks = 0; ks < 8; ++ks) {
        const _Float16* ap = sAh + m * LDH + ks * 32 + hi * 8;
        v8h alo = *(const v8h*)ap;                 // 16B ds_load
        v8h ahi = *(const v8h*)(ap + 16);          // 16B ds_load
        v16h af = __builtin_shufflevector(alo, ahi,
                    0, 1, 2, 3, 4, 5, 6, 7, 8, 9, 10, 11, 12, 13, 14, 15);
        v16h bf = *(const v16h*)(WswAl + ((size_t)(ct * 8 + ks) * 32 + lane) * 16);
        acc = __builtin_amdgcn_wmma_f32_16x16x32_f16(false, af, false, bf,
                                                     (short)0, acc, false, false);
      }
#pragma unroll
      for (int r = 0; r < 8; ++r) {
        float v = acc[r];
        sAlp[(r + hi * 8) * LDH + ct * 16 + n] = v > 0.f ? v : 0.f;
      }
    }
    // fence the async DMA issued in phase 0, then make it WG-visible
    asm volatile("s_wait_asynccnt 0" ::: "memory");
    __syncthreads();
    // phase 3: H1 + AH1 accumulation (thread owns column k)
#pragma unroll
    for (int jj = 0; jj < 16; ++jj) {
      int jg = j0 + jj;
      float h1 = sXi1[k] + sXj1[jj * LDH + k] + sAlp[jj * LDH + k];
      h1 = h1 > 0.f ? h1 : 0.f;
      acc1 = fmaf(sAdj[jg], h1, acc1);
    }
    __syncthreads();
  }
  AH1[(size_t)bi * DD + k] = acc1;
}

// ---------------------------------------------------------------------------
extern "C" void kernel_launch(void* const* d_in, const int* in_sizes, int n_in,
                              void* d_out, int out_size, void* d_ws, size_t ws_size,
                              hipStream_t stream) {
  const float* x          = (const float*)d_in[0];
  const float* rel        = (const float*)d_in[1];
  const float* adj        = (const float*)d_in[2];
  const int*   box_num    = (const int*)d_in[3];
  const float* W_alpha_t  = (const float*)d_in[4];
  const float* b_alpha    = (const float*)d_in[5];
  const float* W_proj     = (const float*)d_in[6];
  const float* b_proj     = (const float*)d_in[7];
  const float* learn_w    = (const float*)d_in[8];
  const float* l0_w_alpha = (const float*)d_in[9];
  const float* l0_w_vi    = (const float*)d_in[10];
  const float* l0_w_vj    = (const float*)d_in[11];
  const float* l0_bias_h  = (const float*)d_in[12];
  const float* l0_w_node  = (const float*)d_in[13];
  // d_in[14] = l1_w_alpha: dead (layer-1 new_alpha is discarded) -> skipped
  const float* l1_w_vi    = (const float*)d_in[15];
  const float* l1_w_vj    = (const float*)d_in[16];
  const float* l1_bias_h  = (const float*)d_in[17];
  const float* l1_w_node  = (const float*)d_in[18];

  float* outf   = (float*)d_out;
  float* x2_out = outf;                       // B*N*OUT = 262144
  float* sa_out = outf + 262144;              // B*N*N   = 262144
  float* gl_out = outf + 524288;              // B       = 4

  float* w = (float*)d_ws;
  size_t o = 0;
  float* xhat = w + o; o += (size_t)BB * NB * LL;   // 131072
  float* adj2 = w + o; o += (size_t)BB * NB * NB;   // 262144
  float* xib0 = w + o; o += (size_t)BB * NB * DD;   // xi0 + b_alpha
  float* xjb0 = w + o; o += (size_t)BB * NB * DD;   // xj0 + bias_h[j,:]
  float* xi1  = w + o; o += (size_t)BB * NB * DD;
  float* xjb1 = w + o; o += (size_t)BB * NB * DD;   // xj1 + l1_bias_h[j,:]
  float* AH0  = w + o; o += (size_t)BB * NB * DD;
  float* AH1  = w + o; o += (size_t)BB * NB * DD;
  float* x1   = w + o; o += (size_t)BB * NB * DD;
  float* p1   = w + o; o += 1024;
  float* p2   = w + o; o += 1024;
  _Float16* wsw = (_Float16*)(w + o);               // 7 f16 weight images
  const size_t WSZ = (size_t)DD * DD;
  _Float16* wswVi0 = wsw + 0 * WSZ;
  _Float16* wswVj0 = wsw + 1 * WSZ;
  _Float16* wswAl0 = wsw + 2 * WSZ;
  _Float16* wswNd0 = wsw + 3 * WSZ;
  _Float16* wswVi1 = wsw + 4 * WSZ;
  _Float16* wswVj1 = wsw + 5 * WSZ;
  _Float16* wswNd1 = wsw + 6 * WSZ;

  // One-time weight f16 swizzles (tiny; L2-resident afterwards)
  swizzle_k<<<256, 256, 0, stream>>>(l0_w_vi,    wswVi0, DD, DD);
  swizzle_k<<<256, 256, 0, stream>>>(l0_w_vj,    wswVj0, DD, DD);
  swizzle_k<<<256, 256, 0, stream>>>(l0_w_alpha, wswAl0, DD, DD);
  swizzle_k<<<256, 256, 0, stream>>>(l0_w_node,  wswNd0, DD, DD);
  swizzle_k<<<256, 256, 0, stream>>>(l1_w_vi,    wswVi1, DD, DD);
  swizzle_k<<<256, 256, 0, stream>>>(l1_w_vj,    wswVj1, DD, DD);
  swizzle_k<<<256, 256, 0, stream>>>(l1_w_node,  wswNd1, DD, DD);

  // Graph-learning branch (fp32)
  xhat_k<<<BB * NB, 128, 0, stream>>>(x, W_proj, b_proj, xhat);
  softadj_k<<<BB * NB, 256, 0, stream>>>(xhat, adj, box_num, learn_w,
                                         sa_out, adj2, p1, p2);
  gl_final_k<<<BB, 256, 0, stream>>>(p1, p2, box_num, gl_out);

  // GCN stack
  dim3 gg(64, 2);  // M/16 x Nout/128, 8 waves per block
  gemm_f16_wmma<<<gg, 256, 0, stream>>>(x,   wswVi0, xib0, 1024, DD, DD, b_alpha, nullptr, 0);
  gemm_f16_wmma<<<gg, 256, 0, stream>>>(x,   wswVj0, xjb0, 1024, DD, DD, nullptr, l0_bias_h, 0);
  gcn0_ah_k<<<BB * NB, 256, 0, stream>>>(xib0, xjb0, rel, W_alpha_t, adj2, AH0);
  gemm_f16_wmma<<<gg, 256, 0, stream>>>(AH0, wswNd0, x1,   1024, DD, DD, nullptr, nullptr, 1);
  gemm_f16_wmma<<<gg, 256, 0, stream>>>(x1,  wswVi1, xi1,  1024, DD, DD, nullptr, nullptr, 0);
  gemm_f16_wmma<<<gg, 256, 0, stream>>>(x1,  wswVj1, xjb1, 1024, DD, DD, nullptr, l1_bias_h, 0);
  gcn1_fused_k<<<BB * NB, 256, 0, stream>>>(xib0, xjb0, rel, W_alpha_t,
                                            xi1, xjb1, wswAl0, adj2, AH1);
  gemm_f16_wmma<<<gg, 256, 0, stream>>>(AH1, wswNd1, x2_out, 1024, DD, DD, nullptr, nullptr, 1);

  (void)in_sizes; (void)n_in; (void)out_size; (void)ws_size;
}
